// NeighborAttention1D_56762287784292
// MI455X (gfx1250) — compile-verified, matrix operand order checked by reading, not run on hardware
//
#include <hip/hip_runtime.h>
#include <stdint.h>

// ---------------------------------------------------------------------------
// NeighborAttention1D collapsed to: out = (smooth3(x) @ Wv^T + s*bv) @ Wo^T + bo
// Prep: xbar = smooth3(x) -> bf16, Wv/Wo -> bf16   (bandwidth kernels)
// GEMM: TDM (tensor_load_to_lds) stages bf16 tiles into padded LDS rows,
//       double-buffered, feeding v_wmma_f32_16x16x32_bf16.
// ---------------------------------------------------------------------------

typedef __bf16 bf16_t;
typedef __attribute__((ext_vector_type(8)))  bf16_t   bf16x8;
typedef __attribute__((ext_vector_type(16))) bf16_t   bf16x16;
typedef __attribute__((ext_vector_type(8)))  float    f32x8;
typedef __attribute__((ext_vector_type(4)))  uint32_t u32x4;
typedef __attribute__((ext_vector_type(4)))  int      i32x4;
typedef __attribute__((ext_vector_type(8)))  int      i32x8;

#define T_DIM 4096
#define D_DIM 1024
#define M_DIM (8 * 4096)

#define BLK_M 128
#define BLK_N 128
#define BLK_K 32
#define KITERS (D_DIM / BLK_K)     // 32
#define ROWB 80                    // 64B data + 16B pad per 32-bf16 LDS row
#define TILEB (BLK_M * ROWB)       // 10240 B per tile buffer

// ---- f32 -> bf16 (RNE) helpers -------------------------------------------
__device__ __forceinline__ uint16_t cvt_bf16(float a) {
    uint32_t u = __float_as_uint(a);
    u = (u + 0x7FFFu + ((u >> 16) & 1u)) >> 16;
    return (uint16_t)u;
}
__device__ __forceinline__ uint32_t pack_bf16(float a, float b) {
    return (uint32_t)cvt_bf16(a) | ((uint32_t)cvt_bf16(b) << 16);
}

// ---- prep kernel 1: xbar[m,d] = w0*x[m-1]+w1*x[m]+w2*x[m+1] (replicate) ---
__global__ __launch_bounds__(256)
void smooth_kernel(const float* __restrict__ x, const float* __restrict__ nw,
                   uint16_t* __restrict__ xbar)
{
    const int idx = blockIdx.x * 256 + threadIdx.x;     // one per 4 elements
    const int m   = idx >> 8;                           // D/4 = 256 groups/row
    const int c   = (idx & 255) * 4;
    const float w0 = nw[0], w1 = nw[1], w2 = nw[2];

    const int t   = m & (T_DIM - 1);
    const int mdn = m - (t > 0 ? 1 : 0);
    const int mup = m + (t < T_DIM - 1 ? 1 : 0);

    const float4 a0 = *(const float4*)(x + (size_t)mdn * D_DIM + c);
    const float4 a1 = *(const float4*)(x + (size_t)m   * D_DIM + c);
    const float4 a2 = *(const float4*)(x + (size_t)mup * D_DIM + c);

    uint2 r;
    r.x = pack_bf16(w0 * a0.x + w1 * a1.x + w2 * a2.x,
                    w0 * a0.y + w1 * a1.y + w2 * a2.y);
    r.y = pack_bf16(w0 * a0.z + w1 * a1.z + w2 * a2.z,
                    w0 * a0.w + w1 * a1.w + w2 * a2.w);
    *(uint2*)(xbar + (size_t)m * D_DIM + c) = r;
}

// ---- prep kernel 2: f32 -> bf16 weight conversion (Wv then Wo) ------------
__global__ __launch_bounds__(256)
void cvtw_kernel(const float* __restrict__ Wv, const float* __restrict__ Wo,
                 uint16_t* __restrict__ Wv16, uint16_t* __restrict__ Wo16)
{
    const int idx = blockIdx.x * 256 + threadIdx.x;     // one per 4 elements
    const size_t nElem4 = (size_t)D_DIM * D_DIM / 4;    // per matrix
    const float*    src = (idx < (int)nElem4) ? Wv   : Wo;
    uint16_t*       dst = (idx < (int)nElem4) ? Wv16 : Wo16;
    const size_t    off = ((size_t)idx % nElem4) * 4;

    const float4 a = *(const float4*)(src + off);
    uint2 r;
    r.x = pack_bf16(a.x, a.y);
    r.y = pack_bf16(a.z, a.w);
    *(uint2*)(dst + off) = r;
}

// ---- TDM: issue a 2D bf16 tile load (32 cols x 128 rows) into padded LDS --
__device__ __forceinline__ void tdm_issue_tile(uint64_t gaddr, uint32_t ldsOff,
                                               uint32_t tensorRows)
{
    u32x4 g0;
    g0[0] = 1u;                                            // count=1, user D#
    g0[1] = ldsOff;                                        // lds_addr (bytes)
    g0[2] = (uint32_t)gaddr;                               // global_addr[31:0]
    g0[3] = ((uint32_t)(gaddr >> 32) & 0x01FFFFFFu)        // global_addr[56:32]
            | (2u << 30);                                  // type=2 (image)

    i32x8 g1;
    g1[0] = (1 << 16)        // data_size = 2 bytes
          | (1 << 20)        // pad_enable
          | (3 << 22)        // pad_interval: 16 DWORDs (64B) between pads
          | (3 << 25);       // pad_amount:   4 DWORDs (16B)
    g1[1] = (int)(( (uint32_t)D_DIM & 0xFFFFu) << 16);     // tensor_dim0 lo16
    g1[2] = (int)(((uint32_t)D_DIM >> 16)                  // tensor_dim0 hi16
          | ((tensorRows & 0xFFFFu) << 16));               // tensor_dim1 lo16
    g1[3] = (int)(((tensorRows >> 16) & 0xFFFFu)           // tensor_dim1 hi16
          | ((uint32_t)BLK_K << 16));                      // tile_dim0 = 32
    g1[4] = (int)BLK_M;                                    // tile_dim1 = 128
    g1[5] = (int)D_DIM;                                    // dim0_stride lo32
    g1[6] = 0;                                             // stride hi bits
    g1[7] = 0;

    const i32x4 z4 = {0, 0, 0, 0};
    const i32x8 z8 = {0, 0, 0, 0, 0, 0, 0, 0};
    // 6-arg form (clang-23 / therock-10.0 headers): g0, g1, g2, g3, g4, cpol
    __builtin_amdgcn_tensor_load_to_lds(g0, g1, z4, z4, z8, 0);
}

// ---- GEMM: C = A_bf16 @ W_bf16^T + bias ----------------------------------
// FIRST = true : bias scaled by sum(nw), C stored as bf16 (y workspace)
// FIRST = false: bias as-is,             C stored as f32  (final output)
template <bool FIRST>
__global__ __launch_bounds__(256)
void tdm_wmma_gemm(const bf16_t* __restrict__ A,    // [rowsA, D] bf16
                   const bf16_t* __restrict__ Wb,   // [D, D] bf16 (out,in)
                   const float*  __restrict__ bias, // [D] f32
                   const float*  __restrict__ nw,   // [3] (FIRST only)
                   void*         __restrict__ Cout)
{
    __shared__ __align__(16) uint8_t smem[4 * TILEB];  // A0 A1 B0 B1

    const int tid  = threadIdx.x;
    const int lane = tid & 31;
    const int wv   = tid >> 5;

    const int mBase = blockIdx.y * BLK_M;
    const int nBase = blockIdx.x * BLK_N;

    const uint32_t ldsBase = (uint32_t)(uintptr_t)&smem[0];
    const uint64_t gA = (uint64_t)(uintptr_t)A + ((uint64_t)mBase * D_DIM) * 2;
    const uint64_t gB = (uint64_t)(uintptr_t)Wb + ((uint64_t)nBase * D_DIM) * 2;

    f32x8 acc[4][2];
#pragma unroll
    for (int i = 0; i < 4; ++i)
#pragma unroll
        for (int j = 0; j < 2; ++j)
#pragma unroll
            for (int e = 0; e < 8; ++e) acc[i][j][e] = 0.0f;

    const int waveM = (wv & 1) * 64;       // wave tile: 64(M) x 32(N)
    const int waveN = (wv >> 1) * 32;
    const int lrow  = lane & 15;
    const int lsel  = (lane >> 4) * 16;    // K-half byte select per lane

    // prologue: DMA tile 0 into buffer 0
    if (wv == 0) {
        tdm_issue_tile(gA, ldsBase + 0 * TILEB, (uint32_t)M_DIM);
        tdm_issue_tile(gB, ldsBase + 2 * TILEB, (uint32_t)D_DIM);
    }

    for (int it = 0; it < KITERS; ++it) {
        const int buf = it & 1;
        if (wv == 0) {
            if (it + 1 < KITERS) {
                const uint64_t kOff = (uint64_t)(it + 1) * BLK_K * 2;
                tdm_issue_tile(gA + kOff, ldsBase + (buf ^ 1) * TILEB, (uint32_t)M_DIM);
                tdm_issue_tile(gB + kOff, ldsBase + (2 + (buf ^ 1)) * TILEB, (uint32_t)D_DIM);
                __builtin_amdgcn_s_wait_tensorcnt(2);  // tile `it` done, it+1 in flight
            } else {
                __builtin_amdgcn_s_wait_tensorcnt(0);
            }
        }
        __syncthreads();   // tile `it` visible to all waves

        const uint8_t* smA = smem + buf * TILEB;
        const uint8_t* smB = smem + (2 + buf) * TILEB;

        bf16x16 bFrag[2];
#pragma unroll
        for (int ns = 0; ns < 2; ++ns) {
            const uint8_t* p = smB + (waveN + ns * 16 + lrow) * ROWB + lsel;
            bf16x8 lo = *(const bf16x8*)(p);
            bf16x8 hi = *(const bf16x8*)(p + 32);
            bFrag[ns] = __builtin_shufflevector(lo, hi, 0, 1, 2, 3, 4, 5, 6, 7,
                                                        8, 9, 10, 11, 12, 13, 14, 15);
        }
#pragma unroll
        for (int ms = 0; ms < 4; ++ms) {
            const uint8_t* p = smA + (waveM + ms * 16 + lrow) * ROWB + lsel;
            bf16x8 lo = *(const bf16x8*)(p);
            bf16x8 hi = *(const bf16x8*)(p + 32);
            bf16x16 aFrag = __builtin_shufflevector(lo, hi, 0, 1, 2, 3, 4, 5, 6, 7,
                                                            8, 9, 10, 11, 12, 13, 14, 15);
#pragma unroll
            for (int ns = 0; ns < 2; ++ns) {
                acc[ms][ns] = __builtin_amdgcn_wmma_f32_16x16x32_bf16(
                    false, aFrag, false, bFrag[ns], (short)0, acc[ms][ns], false, false);
            }
        }
        __syncthreads();   // everyone done reading buf before TDM overwrites it
    }

    // ---- epilogue ----
    const int mOff = (lane >> 4) * 8;
    float bScale = 1.0f;
    if (FIRST) bScale = nw[0] + nw[1] + nw[2];   // faithful bias scaling

#pragma unroll
    for (int ns = 0; ns < 2; ++ns) {
        const int   n   = nBase + waveN + ns * 16 + lrow;
        const float bvv = bias[n] * bScale;
#pragma unroll
        for (int ms = 0; ms < 4; ++ms) {
#pragma unroll
            for (int r = 0; r < 8; ++r) {
                const int   m = mBase + waveM + ms * 16 + mOff + r;
                const float v = acc[ms][ns][r] + bvv;
                if (FIRST) {
                    ((uint16_t*)Cout)[(size_t)m * D_DIM + n] = cvt_bf16(v);
                } else {
                    ((float*)Cout)[(size_t)m * D_DIM + n] = v;
                }
            }
        }
    }
}

extern "C" void kernel_launch(void* const* d_in, const int* in_sizes, int n_in,
                              void* d_out, int out_size, void* d_ws, size_t ws_size,
                              hipStream_t stream) {
    const float* x  = (const float*)d_in[0];
    const float* Wv = (const float*)d_in[1];
    const float* bv = (const float*)d_in[2];
    const float* Wo = (const float*)d_in[3];
    const float* bo = (const float*)d_in[4];
    const float* nw = (const float*)d_in[5];

    // workspace layout (bytes):
    //   [0,   64M)  xbar bf16 [M, D]
    //   [64M, 128M) y    bf16 [M, D]
    //   [128M,130M) Wv   bf16 [D, D]
    //   [130M,132M) Wo   bf16 [D, D]
    const size_t MD2 = (size_t)M_DIM * D_DIM * 2;   // 64 MB
    const size_t DD2 = (size_t)D_DIM * D_DIM * 2;   // 2 MB
    uint16_t* xbar = (uint16_t*)d_ws;
    uint16_t* y    = (uint16_t*)((uint8_t*)d_ws + MD2);
    uint16_t* Wv16 = (uint16_t*)((uint8_t*)d_ws + 2 * MD2);
    uint16_t* Wo16 = (uint16_t*)((uint8_t*)d_ws + 2 * MD2 + DD2);

    // prep: smoothing + weight conversion (bandwidth-bound)
    smooth_kernel<<<(M_DIM * (D_DIM / 4)) / 256, 256, 0, stream>>>(x, nw, xbar);
    cvtw_kernel<<<(2 * D_DIM * (D_DIM / 4)) / 256, 256, 0, stream>>>(Wv, Wo, Wv16, Wo16);

    // two TDM-fed WMMA GEMMs
    dim3 grid(D_DIM / BLK_N, M_DIM / BLK_M);   // (8, 256)
    tdm_wmma_gemm<true ><<<grid, 256, 0, stream>>>(
        (const bf16_t*)xbar, (const bf16_t*)Wv16, bv, nw, y);
    tdm_wmma_gemm<false><<<grid, 256, 0, stream>>>(
        (const bf16_t*)y, (const bf16_t*)Wo16, bo, nullptr, d_out);
}
